// ExNetBFPF_73967926772630
// MI455X (gfx1250) — compile-verified, matrix operand order checked by reading, not run on hardware
//
#include <hip/hip_runtime.h>
#include <hip/hip_bf16.h>

// ---------------------------------------------------------------------------
// Problem constants (from reference)
// ---------------------------------------------------------------------------
#define NB   8            // batch
#define NMIC 8            // mics (M)
#define NF   257          // freq bins (F)
#define NL   497          // frames (L)
#define FLs  (NF * NL)    // 127729 pixels full-res
#define LNC  62           // noise-only frames
#define TTs  (NL - LNC)   // 435 scan steps
#define FDo  129          // downsampled F
#define LDo  249          // downsampled L
#define SDo  (FDo * LDo)  // 32121 pixels down-res
#define BETA_C 0.995f
#define REFM 3

typedef _Float16 half16 __attribute__((ext_vector_type(16)));
typedef float    floatx8 __attribute__((ext_vector_type(8)));

// ---------------------------------------------------------------------------
// Kernel 1: copy Y into channels [0..16) of the 32-channel CNN input x
// ---------------------------------------------------------------------------
__global__ void copy_y_to_x(const float* __restrict__ Y, float* __restrict__ x) {
  size_t i = (size_t)blockIdx.x * blockDim.x + threadIdx.x;
  size_t total = (size_t)NB * 16 * FLs;
  if (i < total) {
    size_t b = i / ((size_t)16 * FLs);
    size_t r = i - b * (size_t)16 * FLs;
    x[b * (size_t)32 * FLs + r] = Y[i];
  }
}

// ---------------------------------------------------------------------------
// Kernel 2: per-(b,f) PAST-d whitened subspace tracking + RTF.
// One wave32 per (b,f). Lanes 0..7 own mic index m. 8x8 complex Hermitian
// eigendecomposition via cyclic Jacobi in LDS (8 lanes parallel over rows).
// Writes rtf real -> x channels 16..23, imag -> 24..31 (t >= LN), zeros t<LN.
// ---------------------------------------------------------------------------
__global__ __launch_bounds__(32) void pastd_rtf(const float* __restrict__ Y,
                                                float* __restrict__ x) {
  __shared__ float s_Ar[8][8], s_Ai[8][8];
  __shared__ float s_Vr[8][8], s_Vi[8][8];
  __shared__ float s_ISr[8][8], s_ISi[8][8];   // Rn^{-1/2}
  __shared__ float s_SQr[8][8], s_SQi[8][8];   // Rn^{+1/2}
  __shared__ float s_ev[8];

  const int bf = blockIdx.x;
  const int b  = bf / NF;
  const int f  = bf - b * NF;
  const int lane = threadIdx.x & 31;

  const float* Yb = Y + (size_t)b * 16 * FLs + (size_t)f * NL;
  float* xout = x + (size_t)b * 32 * FLs + (size_t)f * NL;

  // ---- noise covariance Rn (8x8 complex Hermitian), 2 entries per lane ----
  for (int rep = 0; rep < 2; ++rep) {
    int idx = lane + rep * 32;          // 0..63
    int m = idx >> 3, n = idx & 7;
    float ar = 0.f, ai = 0.f;
    for (int t = 0; t < LNC; ++t) {
      float ymr = Yb[(size_t)m * FLs + t],  ymi = Yb[(size_t)(m + 8) * FLs + t];
      float ynr = Yb[(size_t)n * FLs + t],  yni = Yb[(size_t)(n + 8) * FLs + t];
      ar += ymr * ynr + ymi * yni;      // y_m * conj(y_n)
      ai += ymi * ynr - ymr * yni;
    }
    ar *= (1.f / (float)LNC);
    ai *= (1.f / (float)LNC);
    if (m == n) { ar += 1e-4f; ai = 0.f; }
    s_Ar[m][n] = ar; s_Ai[m][n] = ai;
    s_Vr[m][n] = (m == n) ? 1.f : 0.f;
    s_Vi[m][n] = 0.f;
  }
  __syncthreads();

  // ---- cyclic Jacobi for complex Hermitian 8x8 ----
  for (int sweep = 0; sweep < 6; ++sweep) {
    for (int p = 0; p < 7; ++p) {
      for (int q = p + 1; q < 8; ++q) {
        float apr = s_Ar[p][q], api = s_Ai[p][q];
        float mag = sqrtf(apr * apr + api * api);
        if (mag > 1e-12f) {               // uniform across wave (LDS value)
          float app = s_Ar[p][p], aqq = s_Ar[q][q];
          float tau = (aqq - app) / (2.f * mag);
          float tt  = ((tau >= 0.f) ? 1.f : -1.f) /
                      (fabsf(tau) + sqrtf(1.f + tau * tau));
          float c = 1.f / sqrtf(1.f + tt * tt);
          float s = tt * c;
          float ur = apr / mag, ui = api / mag;   // u = apq/|apq|
          if (lane < 8) {
            int k = lane;
            // column rotation A <- A*G, G=[[c, -s*conj(u)],[s*u, c]]
            float akpr = s_Ar[k][p], akpi = s_Ai[k][p];
            float akqr = s_Ar[k][q], akqi = s_Ai[k][q];
            float nkpr = c * akpr + s * (ur * akqr - ui * akqi);
            float nkpi = c * akpi + s * (ur * akqi + ui * akqr);
            float nkqr = c * akqr - s * (ur * akpr + ui * akpi);
            float nkqi = c * akqi - s * (ur * akpi - ui * akpr);
            s_Ar[k][p] = nkpr; s_Ai[k][p] = nkpi;
            s_Ar[k][q] = nkqr; s_Ai[k][q] = nkqi;
            float vkpr = s_Vr[k][p], vkpi = s_Vi[k][p];
            float vkqr = s_Vr[k][q], vkqi = s_Vi[k][q];
            float mkpr = c * vkpr + s * (ur * vkqr - ui * vkqi);
            float mkpi = c * vkpi + s * (ur * vkqi + ui * vkqr);
            float mkqr = c * vkqr - s * (ur * vkpr + ui * vkpi);
            float mkqi = c * vkqi - s * (ur * vkpi - ui * vkpr);
            s_Vr[k][p] = mkpr; s_Vi[k][p] = mkpi;
            s_Vr[k][q] = mkqr; s_Vi[k][q] = mkqi;
          }
          __syncthreads();
          if (lane < 8) {
            int k = lane;
            // row rotation A <- G^H * A
            float apkr = s_Ar[p][k], apki = s_Ai[p][k];
            float aqkr = s_Ar[q][k], aqki = s_Ai[q][k];
            float npr = c * apkr + s * (ur * aqkr + ui * aqki);
            float npi = c * apki + s * (ur * aqki - ui * aqkr);
            float nqr = c * aqkr - s * (ur * apkr - ui * apki);
            float nqi = c * aqki - s * (ur * apki + ui * apkr);
            s_Ar[p][k] = npr; s_Ai[p][k] = npi;
            s_Ar[q][k] = nqr; s_Ai[q][k] = nqi;
          }
        }
        __syncthreads();
      }
    }
  }
  if (lane < 8) s_ev[lane] = fmaxf(s_Ar[lane][lane], 1e-8f);
  __syncthreads();

  // ---- Rn^{-1/2} and Rn^{1/2} from eigen pairs, 2 entries per lane ----
  for (int rep = 0; rep < 2; ++rep) {
    int idx = lane + rep * 32;
    int m = idx >> 3, n = idx & 7;
    float ir = 0.f, ii = 0.f, sr = 0.f, si = 0.f;
#pragma unroll
    for (int k = 0; k < 8; ++k) {
      float ev = s_ev[k];
      float gi = 1.f / sqrtf(ev);
      float gs = sqrtf(ev);
      float t1r = s_Vr[m][k] * s_Vr[n][k] + s_Vi[m][k] * s_Vi[n][k];
      float t1i = s_Vi[m][k] * s_Vr[n][k] - s_Vr[m][k] * s_Vi[n][k];
      ir += gi * t1r; ii += gi * t1i;
      sr += gs * t1r; si += gs * t1i;
    }
    s_ISr[m][n] = ir; s_ISi[m][n] = ii;
    s_SQr[m][n] = sr; s_SQi[m][n] = si;
  }
  __syncthreads();

  // ---- zero rtf channels for t < LN ----
  for (int idx = lane; idx < 16 * LNC; idx += 32) {
    int ch = idx / LNC, t = idx - ch * LNC;
    xout[(size_t)(16 + ch) * FLs + t] = 0.f;
  }

  const int m = lane & 7;
  auto whiten = [&](int t, float& yr, float& yi) {
    yr = 0.f; yi = 0.f;
#pragma unroll
    for (int n = 0; n < 8; ++n) {
      float cr = s_ISr[m][n], ci = s_ISi[m][n];
      float xr = Yb[(size_t)n * FLs + t], xi = Yb[(size_t)(n + 8) * FLs + t];
      yr += cr * xr - ci * xi;
      yi += cr * xi + ci * xr;
    }
  };

  // ---- w0 = first whitened frame, normalized ----
  float y0r, y0i;
  whiten(LNC, y0r, y0i);
  float nn = y0r * y0r + y0i * y0i;
  nn += __shfl_xor(nn, 1); nn += __shfl_xor(nn, 2); nn += __shfl_xor(nn, 4);
  float inv = 1.f / (sqrtf(nn) + 1e-8f);
  float wr = y0r * inv, wi = y0i * inv;
  float d = 1.f;

  // ---- sequential PAST-d scan over T frames ----
  for (int t = 0; t < TTs; ++t) {
    float yr, yi;
    whiten(LNC + t, yr, yi);
    float ppr = wr * yr + wi * yi;          // conj(w)*y partials
    float ppi = wr * yi - wi * yr;
    ppr += __shfl_xor(ppr, 1); ppr += __shfl_xor(ppr, 2); ppr += __shfl_xor(ppr, 4);
    ppi += __shfl_xor(ppi, 1); ppi += __shfl_xor(ppi, 2); ppi += __shfl_xor(ppi, 4);
    d = BETA_C * d + (ppr * ppr + ppi * ppi);
    float er = yr - (wr * ppr - wi * ppi);  // e = y - w*p
    float ei = yi - (wr * ppi + wi * ppr);
    float gr = ppr / d, gi2 = -ppi / d;     // conj(p)/d
    wr += er * gr - ei * gi2;
    wi += er * gi2 + ei * gr;
    // a = Rn^{1/2} @ w
    float ar = 0.f, ai = 0.f;
#pragma unroll
    for (int n = 0; n < 8; ++n) {
      float wnr = __shfl(wr, n, 8);
      float wni = __shfl(wi, n, 8);
      float cr = s_SQr[m][n], ci = s_SQi[m][n];
      ar += cr * wnr - ci * wni;
      ai += cr * wni + ci * wnr;
    }
    float rfr = __shfl(ar, REFM, 8) + 1e-6f;
    float rfi = __shfl(ai, REFM, 8);
    float den = rfr * rfr + rfi * rfi + 1e-30f;
    float orr = (ar * rfr + ai * rfi) / den;
    float oii = (ai * rfr - ar * rfi) / den;
    if (lane < 8) {
      xout[(size_t)(16 + m) * FLs + (LNC + t)] = orr;
      xout[(size_t)(24 + m) * FLs + (LNC + t)] = oii;
    }
  }
}

// ---------------------------------------------------------------------------
// Implicit-GEMM conv with WMMA f32_16x16x32_f16 and CDNA5 async global->LDS
// staging (GLOBAL_LOAD_ASYNC_TO_LDS_B32, ASYNCcnt), double-buffered so chunk
// k+1 streams into LDS while chunk k is multiplied.
//   D[cout][pixel] = sum_k W[cout][k] * patch[k][pixel],  k = cin*KH*KH + tap
// Block tile: COUT x 64 pixels; waves = (COUT/16)*4, one 16x16 WMMA tile per
// wave. Tiles are staged raw f32 (async copy can't convert) and converted to
// f16 at fragment-build time. UPS fuses nearest upsample into the gather.
// ACT: 0 = relu, 1 = sigmoid.
// ---------------------------------------------------------------------------
template <int CIN, int COUT, int KH, int STRIDE, bool UPS, int ACT>
__global__ void __launch_bounds__(32 * (COUT / 16) * 4)
conv_wmma(const float* __restrict__ in, const float* __restrict__ wgt,
          const float* __restrict__ bias, float* __restrict__ out,
          int inH, int inW, int outH, int outW,
          int inCT, int outCT, int outChOff) {
  constexpr int K    = CIN * KH * KH;
  constexpr int PAD  = KH / 2;
  constexpr int NW   = (COUT / 16) * 4;
  constexpr int NT   = 32 * NW;
  constexpr int BSTR = 68;           // padded f32 row stride for B tile
  constexpr int ASTR = 33;           // padded f32 row stride for A tile
  constexpr int NCH  = (K + 31) / 32;

  __shared__ float Ast[2][COUT * ASTR];
  __shared__ float Bst[2][32 * BSTR];

  const int bb   = blockIdx.y;
  const int outS = outH * outW;
  const int inS  = inH * inW;
  const int tileP = blockIdx.x * 64;
  const int tid  = threadIdx.x;
  const int lane = tid & 31;
  const int wv   = tid >> 5;
  const int wm   = wv >> 2;    // cout tile index
  const int wn   = wv & 3;     // pixel tile index

  const float* inB = in + (size_t)bb * inCT * inS;

  // issue async global->LDS staging of one 32-wide K chunk into buffer `buf`
  auto stage = [&](int kb, int buf) {
    // weight chunk (COUT x 32): per-lane async gather, zeros via DS store
    for (int idx = tid; idx < COUT * 32; idx += NT) {
      int c = idx >> 5, kl = idx & 31, k = kb + kl;
      if (k < K) {
        unsigned lo = (unsigned)(size_t)&Ast[buf][c * ASTR + kl];
        unsigned long long ga =
            (unsigned long long)(size_t)(wgt + (size_t)c * K + k);
        asm volatile("global_load_async_to_lds_b32 %0, %1, off"
                     :: "v"(lo), "v"(ga) : "memory");
      } else {
        Ast[buf][c * ASTR + kl] = 0.f;
      }
    }
    // im2col patch chunk (32 x 64 pixels)
    for (int idx = tid; idx < 32 * 64; idx += NT) {
      int kl = idx >> 6, pp = idx & 63;
      int k = kb + kl;
      int pix = tileP + pp;
      bool ok = false;
      int cin = 0, ih = 0, iw = 0;
      if (k < K && pix < outS) {
        cin = k / (KH * KH);
        int r = k - cin * (KH * KH);
        int ky = r / KH, kx = r - ky * KH;
        int h = pix / outW, w = pix - h * outW;
        int hh = h * STRIDE + ky - PAD;
        int ww = w * STRIDE + kx - PAD;
        if (UPS) {
          ok = (hh >= 0) && (hh < outH) && (ww >= 0) && (ww < outW);
          if (ok) {
            ih = (int)(((float)hh + 0.5f) * (float)inH / (float)outH);
            iw = (int)(((float)ww + 0.5f) * (float)inW / (float)outW);
            if (ih > inH - 1) ih = inH - 1;
            if (iw > inW - 1) iw = inW - 1;
          }
        } else {
          ok = (hh >= 0) && (hh < inH) && (ww >= 0) && (ww < inW);
          ih = hh; iw = ww;
        }
      }
      if (ok) {
        unsigned lo = (unsigned)(size_t)&Bst[buf][kl * BSTR + pp];
        unsigned long long ga = (unsigned long long)(size_t)(
            inB + (size_t)cin * inS + (size_t)ih * inW + iw);
        asm volatile("global_load_async_to_lds_b32 %0, %1, off"
                     :: "v"(lo), "v"(ga) : "memory");
      } else {
        Bst[buf][kl * BSTR + pp] = 0.f;   // zero-pad path (DS store)
      }
    }
  };

  stage(0, 0);                       // prologue: chunk 0 in flight

  floatx8 acc = {};
  const int rowc = wm * 16 + (lane & 15);
  const int pl   = wn * 16 + (lane & 15);
  const int hi   = (lane < 16) ? 0 : 8;

  for (int c = 0; c < NCH; ++c) {
    // drain this wave's async copies, then block-wide barrier: buffer c&1
    // is now fully resident (all waves waited on their own ASYNCcnt).
    asm volatile("s_wait_asynccnt 0" ::: "memory");
    __syncthreads();

    if (c + 1 < NCH) stage((c + 1) * 32, (c + 1) & 1);   // prefetch next

    const float* Ab = Ast[c & 1];
    const float* Bb = Bst[c & 1];
    half16 a{}, b{};
#pragma unroll
    for (int i = 0; i < 16; ++i) {
      int kl = (i < 8) ? (hi + i) : (16 + hi + (i - 8));
      a[i] = (_Float16)Ab[rowc * ASTR + kl];
      b[i] = (_Float16)Bb[kl * BSTR + pl];
    }
    acc = __builtin_amdgcn_wmma_f32_16x16x32_f16(
        /*neg_a=*/false, a, /*neg_b=*/false, b,
        /*c_mod=*/(short)0, acc, /*reuse_a=*/false, /*reuse_b=*/false);
    // no trailing barrier: next iteration's wait+barrier orders the
    // overwrite of buffer c&1 (staged in iteration c+1) after all reads.
  }

  // ---- epilogue: bias + activation, write f32 ----
  const int nloc = lane & 15;
  const int pix = tileP + wn * 16 + nloc;
  if (pix < outS) {
#pragma unroll
    for (int r = 0; r < 8; ++r) {
      int cl = r + ((lane < 16) ? 0 : 8);
      int c  = wm * 16 + cl;
      float v = acc[r] + bias[c];
      if (ACT == 0) v = fmaxf(v, 0.f);
      else          v = 1.f / (1.f + expf(-v));
      out[((size_t)bb * outCT + outChOff + c) * outS + pix] = v;
    }
  }
}

// ---------------------------------------------------------------------------
// Kernel: skip = e1 * gate  -> channels [32..64) of the cat buffer
// ---------------------------------------------------------------------------
__global__ void skip_mul(const float* __restrict__ e1,
                         const float* __restrict__ gate,
                         float* __restrict__ cat) {
  size_t i = (size_t)blockIdx.x * blockDim.x + threadIdx.x;
  size_t total = (size_t)NB * 32 * FLs;
  if (i < total) {
    size_t b = i / ((size_t)32 * FLs);
    size_t r = i - b * (size_t)32 * FLs;
    cat[b * (size_t)64 * FLs + (size_t)32 * FLs + r] = e1[i] * gate[i];
  }
}

// ---------------------------------------------------------------------------
// Kernel: Wc = W_tc[:8] + i W_tc[8:], X_hat = sum_m conj(Wc)*Yc, W_stage2 = 0
// ---------------------------------------------------------------------------
__global__ void finalize(const float* __restrict__ Wtc,
                         const float* __restrict__ Y,
                         float* __restrict__ Xh, float* __restrict__ Wc,
                         float* __restrict__ W2) {
  size_t i = (size_t)blockIdx.x * blockDim.x + threadIdx.x;
  size_t total = (size_t)NB * FLs;
  if (i < total) {
    size_t b = i / FLs;
    size_t p = i - b * FLs;
    float xr = 0.f, xi = 0.f;
#pragma unroll
    for (int m = 0; m < 8; ++m) {
      float wr = Wtc[((size_t)b * 16 + m) * FLs + p];
      float wi = Wtc[((size_t)b * 16 + 8 + m) * FLs + p];
      float yr = Y[((size_t)b * 16 + m) * FLs + p];
      float yi = Y[((size_t)b * 16 + 8 + m) * FLs + p];
      Wc[(((size_t)b * 8 + m) * FLs + p) * 2 + 0] = wr;
      Wc[(((size_t)b * 8 + m) * FLs + p) * 2 + 1] = wi;
      xr += wr * yr + wi * yi;   // conj(w) * y
      xi += wr * yi - wi * yr;
    }
    Xh[((size_t)b * FLs + p) * 2 + 0] = xr;
    Xh[((size_t)b * FLs + p) * 2 + 1] = xi;
    W2[i] = 0.f;
  }
}

// ---------------------------------------------------------------------------
// Host orchestration
// ---------------------------------------------------------------------------
extern "C" void kernel_launch(void* const* d_in, const int* in_sizes, int n_in,
                              void* d_out, int out_size, void* d_ws,
                              size_t ws_size, hipStream_t stream) {
  (void)in_sizes; (void)n_in; (void)out_size; (void)ws_size;

  const float* Y      = (const float*)d_in[0];
  const float* enc1_w = (const float*)d_in[1];
  const float* enc1_b = (const float*)d_in[2];
  const float* down_w = (const float*)d_in[3];
  const float* down_b = (const float*)d_in[4];
  const float* mid_w  = (const float*)d_in[5];
  const float* mid_b  = (const float*)d_in[6];
  const float* up_w   = (const float*)d_in[7];
  const float* up_b   = (const float*)d_in[8];
  const float* gate_w = (const float*)d_in[9];
  const float* gate_b = (const float*)d_in[10];
  const float* out_w  = (const float*)d_in[11];
  const float* out_b  = (const float*)d_in[12];

  float* out = (float*)d_out;
  float* ws  = (float*)d_ws;

  // workspace layout (floats)
  const size_t n_x   = (size_t)NB * 32 * FLs;
  const size_t n_e1  = (size_t)NB * 32 * FLs;
  const size_t n_d1  = (size_t)NB * 64 * SDo;
  const size_t n_mid = (size_t)NB * 64 * SDo;
  float* x   = ws;
  float* e1  = x + n_x;
  float* d1  = e1 + n_e1;
  float* mid = d1 + n_d1;
  float* cat = mid + n_mid;                      // 64 ch full-res [u ; skip]

  // output layout (floats), in reference return order
  const size_t N_WTC = (size_t)NB * 16 * FLs;
  const size_t N_X   = (size_t)NB * FLs * 2;
  const size_t N_Y   = (size_t)NB * 16 * FLs;
  const size_t N_WC  = (size_t)NB * 8 * FLs * 2;
  const size_t N_W2  = (size_t)NB * FLs;
  float* o_Wtc = out;
  float* o_X   = o_Wtc + N_WTC;
  float* o_Y   = o_X + N_X;
  float* o_Wc  = o_Y + N_Y;
  float* o_W2  = o_Wc + N_WC;
  float* o_G   = o_W2 + N_W2;

  // 1) Y passthrough output (device-to-device async copy, capture-safe)
  hipMemcpyAsync(o_Y, Y, N_Y * sizeof(float), hipMemcpyDeviceToDevice, stream);

  // 2) Y -> x channels [0..16)
  {
    size_t total = (size_t)NB * 16 * FLs;
    int blocks = (int)((total + 255) / 256);
    copy_y_to_x<<<blocks, 256, 0, stream>>>(Y, x);
  }

  // 3) PAST-d subspace tracking + RTF -> x channels [16..32)
  pastd_rtf<<<NB * NF, 32, 0, stream>>>(Y, x);

  // 4) CNN (implicit-GEMM WMMA convs, async double-buffered staging)
  dim3 gFull((FLs + 63) / 64, NB);
  dim3 gDown((SDo + 63) / 64, NB);

  // e1 = relu(conv3x3 32->32) on x
  conv_wmma<32, 32, 3, 1, false, 0><<<gFull, 256, 0, stream>>>(
      x, enc1_w, enc1_b, e1, NF, NL, NF, NL, 32, 32, 0);
  // d1 = relu(conv3x3 stride2 32->64)
  conv_wmma<32, 64, 3, 2, false, 0><<<gDown, 512, 0, stream>>>(
      e1, down_w, down_b, d1, NF, NL, FDo, LDo, 32, 64, 0);
  // mid = relu(conv3x3 64->64)
  conv_wmma<64, 64, 3, 1, false, 0><<<gDown, 512, 0, stream>>>(
      d1, mid_w, mid_b, mid, FDo, LDo, FDo, LDo, 64, 64, 0);
  // u = relu(conv3x3 64->32) on nearest-upsampled mid (fused gather),
  //     written into cat channels [0..32)
  conv_wmma<64, 32, 3, 1, true, 0><<<gFull, 256, 0, stream>>>(
      mid, up_w, up_b, cat, FDo, LDo, NF, NL, 64, 64, 0);
  // gate = sigmoid(conv1x1 32->32) on e1, written straight to output slot
  conv_wmma<32, 32, 1, 1, false, 1><<<gFull, 256, 0, stream>>>(
      e1, gate_w, gate_b, o_G, NF, NL, NF, NL, 32, 32, 0);

  // 5) skip = e1 * gate -> cat channels [32..64)
  {
    size_t total = (size_t)NB * 32 * FLs;
    int blocks = (int)((total + 255) / 256);
    skip_mul<<<blocks, 256, 0, stream>>>(e1, o_G, cat);
  }

  // 6) W_tc = sigmoid(conv3x3 64->16) on cat -> output slot
  conv_wmma<64, 16, 3, 1, false, 1><<<gFull, 128, 0, stream>>>(
      cat, out_w, out_b, o_Wtc, NF, NL, NF, NL, 64, 16, 0);

  // 7) Wc, X_hat, W_stage2
  {
    size_t total = (size_t)NB * FLs;
    int blocks = (int)((total + 255) / 256);
    finalize<<<blocks, 256, 0, stream>>>(o_Wtc, Y, o_X, o_Wc, o_W2);
  }
}
